// AttentiveHead_1898375545101
// MI455X (gfx1250) — matrix operand order repaired and checked by reading, not compiled, not run on hardware
//
#include <hip/hip_runtime.h>

#define RR 3
#define NN 300000
#define HH 256
#define GG 2048

typedef __attribute__((ext_vector_type(16))) __bf16         v16bf;
typedef __attribute__((ext_vector_type(16))) unsigned short v16us;
typedef __attribute__((ext_vector_type(8)))  unsigned short v8us;
typedef __attribute__((ext_vector_type(8)))  float          v8f;

// ---------- helpers ----------
__device__ __forceinline__ float fast_tanh(float x) {
#if __has_builtin(__builtin_amdgcn_tanhf)
  return __builtin_amdgcn_tanhf(x);       // gfx1250 v_tanh_f32 (TRANS, co-executes with WMMA)
#else
  return tanhf(x);
#endif
}
__device__ __forceinline__ float fast_exp(float x) { return __expf(x); }  // v_exp_f32 path

__device__ __forceinline__ unsigned short f2bf(float x) {
  union { float f; unsigned u; } c; c.f = x;
  unsigned u = c.u + 0x7fffu + ((c.u >> 16) & 1u);   // round-to-nearest-even
  return (unsigned short)(u >> 16);
}
// monotone float<->uint key for atomic max
__device__ __forceinline__ unsigned fkey(float x) {
  union { float f; unsigned u; } c; c.f = x;
  return (c.u & 0x80000000u) ? ~c.u : (c.u | 0x80000000u);
}
__device__ __forceinline__ float kdec(unsigned k) {
  union { unsigned u; float f; } c;
  c.u = (k & 0x80000000u) ? (k & 0x7fffffffu) : ~k;
  return c.f;
}
__device__ __forceinline__ v16us pack16(float4 f0, float4 f1, float4 f2, float4 f3) {
  v16us t;
  t[0]=f2bf(f0.x);  t[1]=f2bf(f0.y);  t[2]=f2bf(f0.z);  t[3]=f2bf(f0.w);
  t[4]=f2bf(f1.x);  t[5]=f2bf(f1.y);  t[6]=f2bf(f1.z);  t[7]=f2bf(f1.w);
  t[8]=f2bf(f2.x);  t[9]=f2bf(f2.y);  t[10]=f2bf(f2.z); t[11]=f2bf(f2.w);
  t[12]=f2bf(f3.x); t[13]=f2bf(f3.y); t[14]=f2bf(f3.z); t[15]=f2bf(f3.w);
  return t;
}
__device__ __forceinline__ v16us ldsB16(const unsigned short* p) {
  v8us lo = *(const v8us*)p;
  v8us hi = *(const v8us*)(p + 16);
  return __builtin_shufflevector(lo, hi, 0,1,2,3,4,5,6,7,8,9,10,11,12,13,14,15);
}
__device__ __forceinline__ v8f wmma_bf16(v16us a, v16us b, v8f c) {
  return __builtin_amdgcn_wmma_f32_16x16x32_bf16(
      /*neg_a=*/false, __builtin_bit_cast(v16bf, a),
      /*neg_b=*/false, __builtin_bit_cast(v16bf, b),
      /*c_mod=*/(short)0, c, /*reuse_a=*/false, /*reuse_b=*/false);
}

// ---------- K1: per-node attention score  s = tanh(h@W1+b1)@w2 + b2 ----------
// grid (293, R), block 256, dyn LDS = 256*264*2 bytes (W1^T bf16, padded rows)
__global__ void __launch_bounds__(256) k_score(
    const float* __restrict__ h, const float* __restrict__ W1,
    const float* __restrict__ b1, const float* __restrict__ w2,
    const float* __restrict__ b2, float* __restrict__ s_out)
{
  extern __shared__ unsigned short w1s[];                 // [n=256][k padded 264]
  const int r = blockIdx.y, tid = threadIdx.x;
  const float* W1r = W1 + (size_t)r * HH * HH;
  for (int idx = tid; idx < HH * HH; idx += 256) {        // idx = k*256 + n
    int k = idx >> 8, n = idx & 255;
    w1s[n * 264 + k] = f2bf(W1r[idx]);
  }
  __syncthreads();

  const int lane = tid & 31, wave = tid >> 5;
  const int lr = lane & 15;
  const bool hi = lane >= 16;
  const float* hr = h + (size_t)r * NN * HH;
  float* so = s_out + (size_t)r * NN;
  const float b2v = b2[r];

  #pragma unroll 1
  for (int c = 0; c < 8; ++c) {
    int rowBase = blockIdx.x * 1024 + c * 128 + wave * 16;
    if (rowBase >= NN) break;                             // wave-uniform
    int row = rowBase + lr; if (row >= NN) row = NN - 1;

    // A strip: 16 rows x 256 K, bf16, 8 fragments of 16x32 (held in regs)
    v16us afr[8];
    const float* hp = hr + (size_t)row * HH + (hi ? 8 : 0);
    #pragma unroll
    for (int ko = 0; ko < 8; ++ko) {
      const float4* p = (const float4*)(hp + ko * 32);
      afr[ko] = pack16(p[0], p[1], p[4], p[5]);           // K {0..7} and {16..23} (+8 if hi)
    }

    float sacc[8];
    #pragma unroll
    for (int i = 0; i < 8; ++i) sacc[i] = 0.f;

    #pragma unroll 1
    for (int nt = 0; nt < 16; ++nt) {
      v8f acc = {0.f,0.f,0.f,0.f,0.f,0.f,0.f,0.f};
      const unsigned short* bq = &w1s[(nt * 16 + lr) * 264 + (hi ? 8 : 0)];
      // double-buffered B fragments: overlap ds_load with the WMMA pipe
      v16us bb0 = ldsB16(bq);
      v16us bb1 = ldsB16(bq + 32);
      #pragma unroll
      for (int ko = 0; ko < 8; ++ko) {
        v16us bc = (ko & 1) ? bb1 : bb0;
        if (ko + 2 < 8) {
          if (ko & 1) bb1 = ldsB16(bq + (ko + 2) * 32);
          else        bb0 = ldsB16(bq + (ko + 2) * 32);
        }
        acc = wmma_bf16(afr[ko], bc, acc);
      }
      const float bv = b1[r * HH + nt * 16 + lr];
      const float wv = w2[r * HH + nt * 16 + lr];
      #pragma unroll
      for (int i = 0; i < 8; ++i) sacc[i] += fast_tanh(acc[i] + bv) * wv;
    }
    // reduce across the 16 lanes of each half (rows 0-7 in lo half, 8-15 in hi half)
    #pragma unroll
    for (int i = 0; i < 8; ++i) {
      float v = sacc[i];
      v += __shfl_xor(v, 1, 32);
      v += __shfl_xor(v, 2, 32);
      v += __shfl_xor(v, 4, 32);
      v += __shfl_xor(v, 8, 32);
      sacc[i] = v;
    }
    if (lr == 0) {                                        // lanes 0 and 16 write
      int rb = rowBase + (hi ? 8 : 0);
      #pragma unroll
      for (int i = 0; i < 8; ++i)
        if (rb + i < NN) so[rb + i] = sacc[i] + b2v;      // TAU == 1
    }
  }
}

// ---------- K2: segment softmax stats (grid: (ceil(N/256), R)) ----------
__global__ void k_stats1(const float* __restrict__ s, const int* __restrict__ batch,
                         unsigned* __restrict__ mkey, unsigned* __restrict__ cnt) {
  int n = blockIdx.x * 256 + threadIdx.x;
  if (n >= NN) return;
  int r = blockIdx.y;
  size_t t = (size_t)r * NN + n;
  int g = batch[t];
  atomicMax(&mkey[r * GG + g], fkey(s[t]));
  atomicAdd(&cnt[r * GG + g], 1u);
}
__global__ void k_stats2(const float* __restrict__ s, const int* __restrict__ batch,
                         const unsigned* __restrict__ mkey, float* __restrict__ dsum) {
  int n = blockIdx.x * 256 + threadIdx.x;
  if (n >= NN) return;
  int r = blockIdx.y;
  size_t t = (size_t)r * NN + n;
  int g = batch[t];
  atomicAdd(&dsum[r * GG + g], fast_exp(s[t] - kdec(mkey[r * GG + g])));
}
__global__ void k_alpha(const float* __restrict__ s, const int* __restrict__ batch,
                        const unsigned* __restrict__ mkey, const float* __restrict__ dsum,
                        float* __restrict__ alpha) {
  int n = blockIdx.x * 256 + threadIdx.x;
  if (n >= NN) return;
  int r = blockIdx.y;
  size_t t = (size_t)r * NN + n;
  int g = batch[t];
  alpha[t] = fast_exp(s[t] - kdec(mkey[r * GG + g])) / dsum[r * GG + g];
}

// ---------- K3: segmented pooling over h (sorted batch -> register accum + boundary atomics) ----------
// grid (293, R), block 256 (thread = H column)
__global__ void __launch_bounds__(256) k_pool(
    const float* __restrict__ h, const int* __restrict__ batch,
    const float* __restrict__ alpha, float* __restrict__ sm,
    float* __restrict__ att, unsigned* __restrict__ mxkey)
{
  const int r = blockIdx.y, t = threadIdx.x;
  const int base = blockIdx.x * 1024;
  if (base >= NN) return;
  const int end = (base + 1024 < NN) ? base + 1024 : NN;
  const float* hr = h + (size_t)r * NN * HH;
  const int*   br = batch + (size_t)r * NN;
  const float* ar = alpha + (size_t)r * NN;

  int curG = br[base];
  float smA = 0.f, atA = 0.f; unsigned mxA = 0u;
  for (int n = base; n < end; ++n) {
    int g = br[n];                                        // uniform across block
    if (g != curG) {
      size_t o = ((size_t)(r * GG + curG)) * HH + t;
      atomicAdd(&sm[o], smA); atomicAdd(&att[o], atA); atomicMax(&mxkey[o], mxA);
      smA = 0.f; atA = 0.f; mxA = 0u; curG = g;
    }
    float hv = hr[(size_t)n * HH + t];
    float av = ar[n];
    smA += hv; atA += av * hv;
    unsigned k = fkey(hv); mxA = (mxA > k) ? mxA : k;
  }
  size_t o = ((size_t)(r * GG + curG)) * HH + t;
  atomicAdd(&sm[o], smA); atomicAdd(&att[o], atA); atomicMax(&mxkey[o], mxA);
}

// ---------- K4a: build agg = [sum, mean, max, att]  [R*G, 1024] ----------
__global__ void k_agg(const float* __restrict__ sm, const float* __restrict__ att,
                      const unsigned* __restrict__ mxkey, const unsigned* __restrict__ cnt,
                      float* __restrict__ agg) {
  int t = blockIdx.x * 256 + threadIdx.x;
  if (t >= RR * GG * HH) return;
  int hc = t & 255, rg = t >> 8;
  float c = (float)cnt[rg];
  float smv = sm[t], atv = att[t];
  float mxv = (c > 0.f) ? kdec(mxkey[t]) : 0.f;
  float mean = smv / fmaxf(c, 1.f);
  size_t o = (size_t)rg * 1024;
  agg[o + hc] = smv; agg[o + 256 + hc] = mean;
  agg[o + 512 + hc] = mxv; agg[o + 768 + hc] = atv;
}

// ---------- K4b: rv = agg @ Wp + bp  (WMMA, Wp^T 64-col panel in LDS) ----------
// grid (16, 4, R), block 256, dyn LDS = 64*1032*2 bytes
__global__ void __launch_bounds__(256) k_proj(
    const float* __restrict__ agg, const float* __restrict__ Wp,
    const float* __restrict__ bp, float* __restrict__ rv)
{
  extern __shared__ unsigned short wps[];                 // [nl=64][k padded 1032]
  const int r = blockIdx.z, nb = blockIdx.y, tid = threadIdx.x;
  const float* Wpr = Wp + (size_t)r * 1024 * 256;
  for (int idx = tid; idx < 64 * 1024; idx += 256) {      // coalesced: nl fastest
    int nl = idx & 63, k = idx >> 6;
    wps[nl * 1032 + k] = f2bf(Wpr[k * 256 + nb * 64 + nl]);
  }
  __syncthreads();

  const int lane = tid & 31, wave = tid >> 5;
  const int lr = lane & 15; const bool hi = lane >= 16;
  const int gbase = blockIdx.x * 128 + wave * 16;
  const int grow = gbase + lr;
  const float* ap = agg + ((size_t)(r * GG + grow)) * 1024 + (hi ? 8 : 0);

  v8f C[4];
  #pragma unroll
  for (int j = 0; j < 4; ++j) C[j] = (v8f){0.f,0.f,0.f,0.f,0.f,0.f,0.f,0.f};

  // software-pipelined A prefetch (global) over the K loop
  const float4* p0 = (const float4*)ap;
  float4 f0 = p0[0], f1 = p0[1], f2 = p0[4], f3 = p0[5];

  #pragma unroll 1
  for (int ko = 0; ko < 32; ++ko) {
    float4 g0 = f0, g1 = f1, g2 = f2, g3 = f3;
    if (ko < 31) {
      const float4* q = (const float4*)(ap + (ko + 1) * 32);
      g0 = q[0]; g1 = q[1]; g2 = q[4]; g3 = q[5];
    }
    v16us at = pack16(f0, f1, f2, f3);
    const unsigned short* bq0 = &wps[(0 * 16 + lr) * 1032 + ko * 32 + (hi ? 8 : 0)];
    const unsigned short* bq1 = bq0 + 16 * 1032;
    const unsigned short* bq2 = bq0 + 32 * 1032;
    const unsigned short* bq3 = bq0 + 48 * 1032;
    v16us bt0 = ldsB16(bq0), bt1 = ldsB16(bq1), bt2 = ldsB16(bq2), bt3 = ldsB16(bq3);
    C[0] = wmma_bf16(at, bt0, C[0]);
    C[1] = wmma_bf16(at, bt1, C[1]);
    C[2] = wmma_bf16(at, bt2, C[2]);
    C[3] = wmma_bf16(at, bt3, C[3]);
    f0 = g0; f1 = g1; f2 = g2; f3 = g3;
  }
  const int rowo = gbase + (hi ? 8 : 0);
  #pragma unroll
  for (int j = 0; j < 4; ++j) {
    int col = nb * 64 + j * 16 + lr;
    float bpv = bp[r * 256 + col];
    #pragma unroll
    for (int i = 0; i < 8; ++i)
      rv[((size_t)(r * GG + rowo + i)) * 256 + col] = C[j][i] + bpv;
  }
}

// ---------- K5: final MLP  LN -> SiLU -> Linear -> SiLU -> Linear ----------
#define GPB 8
__global__ void __launch_bounds__(256) k_final(
    const float* __restrict__ rv, const float* __restrict__ lng,
    const float* __restrict__ lnb, const float* __restrict__ Wf1,
    const float* __restrict__ bf1, const float* __restrict__ Wf2,
    const float* __restrict__ bf2, float* __restrict__ out)
{
  __shared__ float xs[GPB][768];
  __shared__ float red[256];
  __shared__ float s_mu[GPB], s_inv[GPB];
  const int g0 = blockIdx.x * GPB, t = threadIdx.x;

  float v[GPB][3];
  #pragma unroll
  for (int q = 0; q < GPB; ++q)
    #pragma unroll
    for (int r3 = 0; r3 < 3; ++r3) {
      float x = rv[((size_t)(r3 * GG + g0 + q)) * HH + t];
      v[q][r3] = x; xs[q][r3 * 256 + t] = x;
    }
  __syncthreads();

  for (int q = 0; q < GPB; ++q) {
    red[t] = v[q][0] + v[q][1] + v[q][2];
    __syncthreads();
    for (int off = 128; off > 0; off >>= 1) { if (t < off) red[t] += red[t + off]; __syncthreads(); }
    if (t == 0) s_mu[q] = red[0] * (1.f / 768.f);
    __syncthreads();
    float mu = s_mu[q];
    float d0 = v[q][0] - mu, d1 = v[q][1] - mu, d2 = v[q][2] - mu;
    red[t] = d0 * d0 + d1 * d1 + d2 * d2;
    __syncthreads();
    for (int off = 128; off > 0; off >>= 1) { if (t < off) red[t] += red[t + off]; __syncthreads(); }
    if (t == 0) s_inv[q] = rsqrtf(red[0] * (1.f / 768.f) + 1e-5f);
    __syncthreads();
  }

  #pragma unroll
  for (int q = 0; q < GPB; ++q) {
    float mu = s_mu[q], inv = s_inv[q];
    #pragma unroll
    for (int r3 = 0; r3 < 3; ++r3) {
      int idx = r3 * 256 + t;
      float xn = (xs[q][idx] - mu) * inv * lng[idx] + lnb[idx];
      xs[q][idx] = xn / (1.f + fast_exp(-xn));            // SiLU
    }
  }
  __syncthreads();

  float acc[GPB];
  #pragma unroll
  for (int q = 0; q < GPB; ++q) acc[q] = bf1[t];
  for (int k = 0; k < 768; ++k) {
    float wv = Wf1[k * 256 + t];
    #pragma unroll
    for (int q = 0; q < GPB; ++q) acc[q] += xs[q][k] * wv;
  }
  const float w2v = Wf2[t], b2v = bf2[0];
  for (int q = 0; q < GPB; ++q) {
    float y = acc[q] / (1.f + fast_exp(-acc[q]));
    red[t] = y * w2v;
    __syncthreads();
    for (int off = 128; off > 0; off >>= 1) { if (t < off) red[t] += red[t + off]; __syncthreads(); }
    if (t == 0) out[g0 + q] = red[0] + b2v;
    __syncthreads();
  }
}

// ---------- workspace layout (in 4-byte words) ----------
enum : size_t {
  OFF_S    = 0,                                // s      [R*N]
  OFF_A    = 900000,                           // alpha  [R*N]
  OFF_MKEY = 1800000,                          // mkey   [R*G] u32
  OFF_D    = 1806144,                          // denom  [R*G]
  OFF_CNT  = 1812288,                          // cnt    [R*G] u32
  OFF_SM   = 1818432,                          // sum    [R*G*H]
  OFF_ATT  = 3391296,                          // att    [R*G*H]
  OFF_MXK  = 4964160,                          // maxkey [R*G*H] u32
  OFF_AGG  = 6537024,                          // agg    [R*G*4H]
  OFF_RV   = 12828480,                         // rv     [R*G*H]
  WS_WORDS = 14401344
};

extern "C" void kernel_launch(void* const* d_in, const int* in_sizes, int n_in,
                              void* d_out, int out_size, void* d_ws, size_t ws_size,
                              hipStream_t stream) {
  (void)in_sizes; (void)n_in; (void)out_size;
  const float* h   = (const float*)d_in[0];
  const int*   bat = (const int*)  d_in[1];
  const float* W1  = (const float*)d_in[2];
  const float* b1  = (const float*)d_in[3];
  const float* w2  = (const float*)d_in[4];
  const float* b2  = (const float*)d_in[5];
  const float* Wp  = (const float*)d_in[6];
  const float* bp  = (const float*)d_in[7];
  const float* lng = (const float*)d_in[8];
  const float* lnb = (const float*)d_in[9];
  const float* Wf1 = (const float*)d_in[10];
  const float* bf1 = (const float*)d_in[11];
  const float* Wf2 = (const float*)d_in[12];
  const float* bf2 = (const float*)d_in[13];
  float* out = (float*)d_out;
  float* ws  = (float*)d_ws;
  if (ws_size < WS_WORDS * sizeof(float)) return;         // need ~57.6 MB scratch

  float*    s     = ws + OFF_S;
  float*    alpha = ws + OFF_A;
  unsigned* mkey  = (unsigned*)(ws + OFF_MKEY);
  float*    dsum  = ws + OFF_D;
  unsigned* cnt   = (unsigned*)(ws + OFF_CNT);
  float*    sm    = ws + OFF_SM;
  float*    att   = ws + OFF_ATT;
  unsigned* mxk   = (unsigned*)(ws + OFF_MXK);
  float*    agg   = ws + OFF_AGG;
  float*    rv    = ws + OFF_RV;

  const int K1_LDS = 256 * 264 * 2;                       // 135168 B (<=320KB/WGP)
  const int K4_LDS = 64 * 1032 * 2;                       // 132096 B
  (void)hipFuncSetAttribute((const void*)k_score, hipFuncAttributeMaxDynamicSharedMemorySize, K1_LDS);
  (void)hipFuncSetAttribute((const void*)k_proj,  hipFuncAttributeMaxDynamicSharedMemorySize, K4_LDS);

  // zero the accumulated regions (mkey..mxkey end) every call
  hipMemsetAsync(ws + OFF_MKEY, 0, (OFF_AGG - OFF_MKEY) * sizeof(float), stream);

  const int rowBlocks = (NN + 1023) / 1024;               // 293
  k_score<<<dim3(rowBlocks, RR), 256, K1_LDS, stream>>>(h, W1, b1, w2, b2, s);

  const int nBlk = (NN + 255) / 256;
  k_stats1<<<dim3(nBlk, RR), 256, 0, stream>>>(s, bat, mkey, cnt);
  k_stats2<<<dim3(nBlk, RR), 256, 0, stream>>>(s, bat, mkey, dsum);
  k_alpha <<<dim3(nBlk, RR), 256, 0, stream>>>(s, bat, mkey, dsum, alpha);

  k_pool<<<dim3(rowBlocks, RR), 256, 0, stream>>>(h, bat, alpha, sm, att, mxk);

  k_agg<<<(RR * GG * HH + 255) / 256, 256, 0, stream>>>(sm, att, mxk, cnt, agg);

  k_proj<<<dim3(16, 4, RR), 256, K4_LDS, stream>>>(agg, Wp, bp, rv);

  k_final<<<GG / GPB, 256, 0, stream>>>(rv, lng, lnb, Wf1, bf1, Wf2, bf2, out);
}
